// _TransformerBlock_50010599194883
// MI455X (gfx1250) — matrix operand order repaired and checked by reading, not compiled, not run on hardware
//
#include <hip/hip_runtime.h>

typedef unsigned int u32;
typedef __attribute__((ext_vector_type(16))) __bf16 v16bf;
typedef __attribute__((ext_vector_type(8)))  float  v8f;

// Async global->LDS staging (CDNA5 GLOBAL_LOAD_ASYNC_TO_LDS_B128, ASYNCcnt).
// Set to 0 to fall back to the VGPR-staged synchronous path.
#define USE_ASYNC_COPY 1

// Low 32 bits of a generic pointer into the LDS aperture are the LDS byte
// offset (ISA 10.2).
__device__ __forceinline__ void async_cp16(const void* gsrc, void* lds_dst) {
#if USE_ASYNC_COPY
  u32 ldsaddr = (u32)(size_t)lds_dst;
  asm volatile("global_load_async_to_lds_b128 %0, %1, off"
               :: "v"(ldsaddr), "v"(gsrc) : "memory");
#endif
}
__device__ __forceinline__ void wait_async0() {
#if USE_ASYNC_COPY
  asm volatile("s_wait_asynccnt 0" ::: "memory");
#endif
}

// ---------------------------------------------------------------------------
// WMMA fragment index helpers (CDNA5 ISA 7.12.2, 16-bit operands, wave32)
// A-matrix 16x32: lane%16 = M row; element e=2v+s -> K = 16*(v>>2)+8*(lane/16)+2*(v&3)+s
// B-matrix 32x16: lane%16 = N col; element e=2v+s -> K = 16*(v>>2)+8*(lane/16)+2*(v&3)+s
// C/D 16x16 f32 : lane%16 = N;     vgpr j -> M = j + 8*(lane/16)
// ---------------------------------------------------------------------------
__device__ __forceinline__ int kA_idx(int v, int half) { return 16 * (v >> 2) + 8 * half + 2 * (v & 3); }
__device__ __forceinline__ int kB_idx(int v, int half) { return 16 * (v >> 2) + 8 * half + 2 * (v & 3); }

__device__ __forceinline__ v16bf frag_a_lds(const u32* lds, int row0, int lane16, int half) {
  union { v16bf v; u32 u[8]; } f;
  int m = row0 + lane16;
#pragma unroll
  for (int t = 0; t < 8; ++t) f.u[t] = lds[m * 16 + (kA_idx(t, half) >> 1)];
  return f.v;
}
__device__ __forceinline__ v16bf frag_b_lds(const u32* lds, int row0, int lane16, int half) {
  union { v16bf v; u32 u[8]; } f;
  int n = row0 + lane16;
#pragma unroll
  for (int t = 0; t < 8; ++t) f.u[t] = lds[n * 16 + (kB_idx(t, half) >> 1)];
  return f.v;
}

__device__ __forceinline__ float gelu_exact(float v) {
  return 0.5f * v * (1.0f + erff(v * 0.70710678118654752f));
}

// ---------------------------------------------------------------------------
// Weight convert + transpose: W[K][N] f32 -> Wt[N][K] bf16
// ---------------------------------------------------------------------------
__global__ void convt_kernel(const float* __restrict__ W, __bf16* __restrict__ Wt, int K, int N) {
  int idx = blockIdx.x * 256 + threadIdx.x;
  if (idx >= K * N) return;
  int n = idx / K, k = idx - n * K;
  Wt[idx] = (__bf16)W[(size_t)k * N + n];
}

// ---------------------------------------------------------------------------
// LayerNorm over rows of 512, fp32 in -> bf16 out. 1 block / row, 128 thr.
// ---------------------------------------------------------------------------
__global__ __launch_bounds__(128) void layernorm_kernel(
    const float* __restrict__ x, const float* __restrict__ g,
    const float* __restrict__ b, __bf16* __restrict__ out) {
  __shared__ float ws_[4], wss_[4];
  const int row = blockIdx.x, tid = threadIdx.x;
  float4 v = ((const float4*)(x + (size_t)row * 512))[tid];
  float s = v.x + v.y + v.z + v.w;
  float ss = v.x * v.x + v.y * v.y + v.z * v.z + v.w * v.w;
#pragma unroll
  for (int off = 1; off < 32; off <<= 1) {
    s += __shfl_xor(s, off, 32);
    ss += __shfl_xor(ss, off, 32);
  }
  if ((tid & 31) == 0) { ws_[tid >> 5] = s; wss_[tid >> 5] = ss; }
  __syncthreads();
  s = ws_[0] + ws_[1] + ws_[2] + ws_[3];
  ss = wss_[0] + wss_[1] + wss_[2] + wss_[3];
  float mu = s * (1.0f / 512.0f);
  float rstd = rsqrtf(ss * (1.0f / 512.0f) - mu * mu + 1e-5f);
  float4 gv = ((const float4*)g)[tid];
  float4 bv = ((const float4*)b)[tid];
  union { __bf16 h[4]; uint2 u2; } pk;
  pk.h[0] = (__bf16)((v.x - mu) * rstd * gv.x + bv.x);
  pk.h[1] = (__bf16)((v.y - mu) * rstd * gv.y + bv.y);
  pk.h[2] = (__bf16)((v.z - mu) * rstd * gv.z + bv.z);
  pk.h[3] = (__bf16)((v.w - mu) * rstd * gv.w + bv.w);
  ((uint2*)(out + (size_t)row * 512))[tid] = pk.u2;
}

// ---------------------------------------------------------------------------
// BF16 WMMA GEMM:  C[M,N] = A[M,K](bf16,row) * Bt[N,K](bf16,row)^T
// Block tile 128x128, BK=32, 256 threads = 8 waves, each wave 32x64 (2x4 WMMA).
// Tiles staged with async global->LDS b128 copies (no VGPR round-trip).
// Epilogues: 0 f32 | 1 bf16 | 2 +res f32 | 3 gelu bf16 | 4 qkv split
// ---------------------------------------------------------------------------
enum { EPI_F32 = 0, EPI_BF16 = 1, EPI_RES_F32 = 2, EPI_GELU_BF16 = 3, EPI_QKV = 4 };

template <int EPI>
__global__ __launch_bounds__(256) void gemm_bf16(
    const __bf16* __restrict__ A, const __bf16* __restrict__ Bt,
    int M, int N, int K,
    const float* __restrict__ res, float* __restrict__ outF,
    __bf16* __restrict__ outB, __bf16* __restrict__ outB2) {
  __shared__ u32 lds_a[128 * 16];  // 128 x 32 bf16 (8 KB)
  __shared__ u32 lds_b[128 * 16];  // 128 x 32 bf16 (8 KB)
  const int tid = threadIdx.x, lane = tid & 31, wave = tid >> 5;
  const int wm = wave >> 1;        // 0..3 -> rows wm*32
  const int wn = wave & 1;         // 0..1 -> cols wn*64
  const int lane16 = lane & 15, half = lane >> 4;
  const int m0 = blockIdx.y * 128, n0 = blockIdx.x * 128;
  const int Ku = K >> 1;

  v8f acc[2][4] = {};
  for (int kt = 0; kt < K; kt += 32) {
    __syncthreads();
#if USE_ASYNC_COPY
    // Each 128x32 bf16 tile = 512 x 16B chunks; 2 chunks / thread / matrix.
#pragma unroll
    for (int i = 0; i < 2; ++i) {
      int c16 = tid + i * 256;
      int r = c16 >> 2, c = c16 & 3;  // c = 16B unit within a 64B row
      async_cp16(A + (size_t)(m0 + r) * K + kt + c * 8, &lds_a[r * 16 + c * 4]);
    }
#pragma unroll
    for (int i = 0; i < 2; ++i) {
      int c16 = tid + i * 256;
      int r = c16 >> 2, c = c16 & 3;
      async_cp16(Bt + (size_t)(n0 + r) * K + kt + c * 8, &lds_b[r * 16 + c * 4]);
    }
    wait_async0();
#else
    const u32* Au = (const u32*)A;
    const u32* Bu = (const u32*)Bt;
#pragma unroll
    for (int i = 0; i < 8; ++i) {  // A tile: 2048 dwords
      int idx = tid + i * 256, r = idx >> 4, c = idx & 15;
      lds_a[idx] = Au[(size_t)(m0 + r) * Ku + (kt >> 1) + c];
    }
#pragma unroll
    for (int i = 0; i < 8; ++i) {  // B tile: 2048 dwords
      int idx = tid + i * 256, r = idx >> 4, c = idx & 15;
      lds_b[idx] = Bu[(size_t)(n0 + r) * Ku + (kt >> 1) + c];
    }
#endif
    __syncthreads();
    v16bf af[2], bfr[4];
#pragma unroll
    for (int i = 0; i < 2; ++i) af[i] = frag_a_lds(lds_a, wm * 32 + i * 16, lane16, half);
#pragma unroll
    for (int j = 0; j < 4; ++j) bfr[j] = frag_b_lds(lds_b, wn * 64 + j * 16, lane16, half);
#pragma unroll
    for (int i = 0; i < 2; ++i)
#pragma unroll
      for (int j = 0; j < 4; ++j)
        acc[i][j] = __builtin_amdgcn_wmma_f32_16x16x32_bf16(false, af[i], false, bfr[j], (short)0, acc[i][j], false, false);
  }
  (void)Ku;
#pragma unroll
  for (int i = 0; i < 2; ++i)
#pragma unroll
    for (int j = 0; j < 4; ++j)
#pragma unroll
      for (int t = 0; t < 8; ++t) {
        int row = m0 + wm * 32 + i * 16 + t + 8 * half;
        int col = n0 + wn * 64 + j * 16 + lane16;
        float v = acc[i][j][t];
        if constexpr (EPI == EPI_F32) {
          outF[(size_t)row * N + col] = v;
        } else if constexpr (EPI == EPI_BF16) {
          outB[(size_t)row * N + col] = (__bf16)v;
        } else if constexpr (EPI == EPI_RES_F32) {
          outF[(size_t)row * N + col] = v + res[(size_t)row * N + col];
        } else if constexpr (EPI == EPI_GELU_BF16) {
          outB[(size_t)row * N + col] = (__bf16)gelu_exact(v);
        } else {  // EPI_QKV: N==1536, q/k rows to outB, v transposed to outB2[B,H,Dh,T]
          if (col < 1024) {
            outB[(size_t)row * 1536 + col] = (__bf16)v;
          } else {
            int c2 = col - 1024, hh = c2 >> 7, dd = c2 & 127;
            int bb = row >> 11, t2 = row & 2047;
            outB2[((size_t)((bb << 2) + hh) * 128 + dd) * 2048 + t2] = (__bf16)v;
          }
        }
      }
}

// ---------------------------------------------------------------------------
// Causal flash attention. One wave per 16-query tile of one (b,h).
// Q fragments are pre-scaled by 1/sqrt(Dh); tiles strictly below the diagonal
// take the MASKED=false path (no compares/cndmask in the hot loop).
// ---------------------------------------------------------------------------
template <bool MASKED>
__device__ __forceinline__ void attn_tile(
    int kbase, int qbase, int lane16, int half,
    const u32* __restrict__ kpu, const u32* __restrict__ vpu, float* lp,
    const v16bf* qv, v8f* o, float* rmax, float* rsum) {
  // ---- scores S[16q x 32k] = Q * K^T : two N-groups of 16 keys ----
  v8f s[2] = {};
#pragma unroll
  for (int g = 0; g < 2; ++g) {
    int krow = kbase + g * 16 + lane16;  // key = B-matrix column
#pragma unroll
    for (int kc = 0; kc < 4; ++kc) {
      union { v16bf v; u32 u[8]; } kf;
#pragma unroll
      for (int t = 0; t < 8; ++t) {
        int d = kc * 32 + kB_idx(t, half);
        kf.u[t] = kpu[(size_t)krow * 768 + (d >> 1)];
      }
      s[g] = __builtin_amdgcn_wmma_f32_16x16x32_bf16(false, qv[kc], false, kf.v, (short)0, s[g], false, false);
    }
  }
  if constexpr (MASKED) {
#pragma unroll
    for (int g = 0; g < 2; ++g) {
      int kg = kbase + g * 16 + lane16;
#pragma unroll
      for (int j = 0; j < 8; ++j) {
        int qg = qbase + j + 8 * half;
        if (kg > qg) s[g][j] = -3.0e38f;
      }
    }
  }
  // ---- online softmax (row = (j, half) group of 16 lanes) ----
  float p0[8], p1[8];
#pragma unroll
  for (int j = 0; j < 8; ++j) {
    float tmax = fmaxf(s[0][j], s[1][j]);
#pragma unroll
    for (int off = 1; off < 16; off <<= 1) tmax = fmaxf(tmax, __shfl_xor(tmax, off, 32));
    float nm = fmaxf(rmax[j], tmax);
    float cf = __expf(rmax[j] - nm);
    rmax[j] = nm;
    float e0 = __expf(s[0][j] - nm);
    float e1 = __expf(s[1][j] - nm);
    p0[j] = e0; p1[j] = e1;
    float rs = e0 + e1;
#pragma unroll
    for (int off = 1; off < 16; off <<= 1) rs += __shfl_xor(rs, off, 32);
    rsum[j] = rsum[j] * cf + rs;
#pragma unroll
    for (int f = 0; f < 8; ++f) o[f][j] *= cf;
  }
  // ---- relayout P (C/D layout) -> A-fragment via per-wave LDS scratch ----
#pragma unroll
  for (int j = 0; j < 8; ++j) {
    int m = j + 8 * half;
    lp[m * 32 + lane16] = p0[j];
    lp[m * 32 + 16 + lane16] = p1[j];
  }
  union { v16bf v; u32 u[8]; } pf;
#pragma unroll
  for (int t = 0; t < 8; ++t) {
    int kk = kA_idx(t, half);
    union { u32 u; __bf16 hh[2]; } pk;
    pk.hh[0] = (__bf16)lp[lane16 * 32 + kk];
    pk.hh[1] = (__bf16)lp[lane16 * 32 + kk + 1];
    pf.u[t] = pk.u;
  }
  // ---- O[16q x 128d] += P[16x32] * V[32k x 128d] (V^T layout: contiguous keys) ----
#pragma unroll
  for (int f = 0; f < 8; ++f) {
    int dh = f * 16 + lane16;  // B-matrix column = dh
    union { v16bf v; u32 u[8]; } vf;
#pragma unroll
    for (int t = 0; t < 8; ++t) {
      int kk = kbase + kB_idx(t, half);
      vf.u[t] = vpu[(size_t)dh * 1024 + (kk >> 1)];
    }
    o[f] = __builtin_amdgcn_wmma_f32_16x16x32_bf16(false, pf.v, false, vf.v, (short)0, o[f], false, false);
  }
}

__global__ __launch_bounds__(256) void attn_kernel(
    const __bf16* __restrict__ qkv, const __bf16* __restrict__ vT,
    __bf16* __restrict__ out) {
  __shared__ float lds_p[8][16 * 32];
  const int T = 2048;
  const int lane = threadIdx.x & 31, wave = threadIdx.x >> 5;
  const int task = blockIdx.x * 8 + wave;       // B*H*(T/16) = 2048 tasks
  const int qt = task & 127;                    // T/16 = 128
  const int h = (task >> 7) & 3;
  const int b = task >> 9;
  const int qbase = qt * 16;
  const int lane16 = lane & 15, half = lane >> 4;
  const float scale = 0.08838834764831845f;     // 1/sqrt(128)

  const u32* qpu = (const u32*)(qkv + (size_t)b * T * 1536 + (size_t)h * 128);
  const u32* kpu = qpu + 256;                   // +512 bf16
  const u32* vpu = (const u32*)(vT + (size_t)(b * 4 + h) * 128 * T);

  // Q fragments (4 K-chunks of 32 over Dh=128), pre-scaled by 1/sqrt(Dh)
  union { v16bf v; u32 u[8]; } qf[4];
  const int qrow = qbase + lane16;
#pragma unroll
  for (int kc = 0; kc < 4; ++kc)
#pragma unroll
    for (int t = 0; t < 8; ++t) {
      int d = kc * 32 + kA_idx(t, half);
      union { u32 u; __bf16 hh[2]; } pk;
      pk.u = qpu[(size_t)qrow * 768 + (d >> 1)];
      pk.hh[0] = (__bf16)((float)pk.hh[0] * scale);
      pk.hh[1] = (__bf16)((float)pk.hh[1] * scale);
      qf[kc].u[t] = pk.u;
    }
  v16bf qv[4];
#pragma unroll
  for (int kc = 0; kc < 4; ++kc) qv[kc] = qf[kc].v;

  v8f o[8] = {};
  float rmax[8], rsum[8];
#pragma unroll
  for (int j = 0; j < 8; ++j) { rmax[j] = -3.0e38f; rsum[j] = 0.0f; }

  float* lp = lds_p[wave];
  const int nfull = qbase >> 5;  // tiles strictly below the diagonal
  for (int kb = 0; kb < nfull; ++kb)
    attn_tile<false>(kb * 32, qbase, lane16, half, kpu, vpu, lp, qv, o, rmax, rsum);
  attn_tile<true>(nfull * 32, qbase, lane16, half, kpu, vpu, lp, qv, o, rmax, rsum);

  // ---- finalize: divide by row sums, store bf16 token-major ----
  __bf16* op = out + (size_t)b * T * 512 + (size_t)h * 128;
#pragma unroll
  for (int j = 0; j < 8; ++j) {
    float inv = 1.0f / rsum[j];
    int row = qbase + j + 8 * half;
#pragma unroll
    for (int f = 0; f < 8; ++f)
      op[(size_t)row * 512 + f * 16 + lane16] = (__bf16)(o[f][j] * inv);
  }
}

// ---------------------------------------------------------------------------
// Host: orchestrate the full block.
// ---------------------------------------------------------------------------
extern "C" void kernel_launch(void* const* d_in, const int* in_sizes, int n_in,
                              void* d_out, int out_size, void* d_ws, size_t ws_size,
                              hipStream_t stream) {
  const float* x      = (const float*)d_in[0];
  const float* ln1_g  = (const float*)d_in[1];
  const float* ln1_b  = (const float*)d_in[2];
  const float* w_qkv  = (const float*)d_in[3];
  const float* w_proj = (const float*)d_in[4];
  const float* ln2_g  = (const float*)d_in[5];
  const float* ln2_b  = (const float*)d_in[6];
  const float* w_ff1  = (const float*)d_in[7];
  const float* w_ff2  = (const float*)d_in[8];
  float* out = (float*)d_out;

  const int B = 4, T = 2048, C = 512, H = 4, FF = 2048;
  const int M = B * T;  // 8192 tokens

  char* ws = (char*)d_ws;
  size_t off = 0;
  auto alloc = [&](size_t bytes) -> void* {
    void* p = ws + off;
    off += (bytes + 255) & ~(size_t)255;
    return p;
  };
  __bf16* wqkvT = (__bf16*)alloc((size_t)(3 * C) * C * 2);   // [1536,512]
  __bf16* wprojT = (__bf16*)alloc((size_t)C * C * 2);        // [512,512]
  __bf16* wff1T = (__bf16*)alloc((size_t)FF * C * 2);        // [2048,512]
  __bf16* wff2T = (__bf16*)alloc((size_t)C * FF * 2);        // [512,2048]
  __bf16* bigA = (__bf16*)alloc((size_t)M * FF * 2);         // qkv (M x 1536) then gelu acts (M x 2048)
  __bf16* vT = (__bf16*)alloc((size_t)B * C * T * 2);        // [B,H,128,T]
  __bf16* xbuf = (__bf16*)alloc((size_t)M * C * 2);          // xn1 -> attn_out -> xn2
  (void)ws_size; (void)in_sizes; (void)n_in; (void)out_size;

  // 1) weights -> bf16, transposed [N,K]
  convt_kernel<<<(C * 3 * C + 255) / 256, 256, 0, stream>>>(w_qkv, wqkvT, C, 3 * C);
  convt_kernel<<<(C * C + 255) / 256, 256, 0, stream>>>(w_proj, wprojT, C, C);
  convt_kernel<<<(C * FF + 255) / 256, 256, 0, stream>>>(w_ff1, wff1T, C, FF);
  convt_kernel<<<(FF * C + 255) / 256, 256, 0, stream>>>(w_ff2, wff2T, FF, C);

  // 2) LN1
  layernorm_kernel<<<M, 128, 0, stream>>>(x, ln1_g, ln1_b, xbuf);

  // 3) QKV GEMM (q,k packed rows; v scattered transposed)
  gemm_bf16<EPI_QKV><<<dim3(3 * C / 128, M / 128), 256, 0, stream>>>(
      xbuf, wqkvT, M, 3 * C, C, nullptr, nullptr, bigA, vT);

  // 4) causal flash attention -> xbuf (attn_out, bf16, token-major)
  attn_kernel<<<(B * H * (T / 16)) / 8, 256, 0, stream>>>(bigA, vT, xbuf);

  // 5) proj GEMM + residual x -> d_out (this is x2)
  gemm_bf16<EPI_RES_F32><<<dim3(C / 128, M / 128), 256, 0, stream>>>(
      xbuf, wprojT, M, C, C, x, out, nullptr, nullptr);

  // 6) LN2 on x2
  layernorm_kernel<<<M, 128, 0, stream>>>(out, ln2_g, ln2_b, xbuf);

  // 7) FF1 GEMM + exact GELU -> bigA
  gemm_bf16<EPI_GELU_BF16><<<dim3(FF / 128, M / 128), 256, 0, stream>>>(
      xbuf, wff1T, M, FF, C, nullptr, nullptr, bigA, nullptr);

  // 8) FF2 GEMM + residual x2 -> d_out (same-element read-then-write per lane)
  gemm_bf16<EPI_RES_F32><<<dim3(C / 128, M / 128), 256, 0, stream>>>(
      bigA, wff2T, M, C, FF, out, out, nullptr, nullptr);
}
